// ContextPrior3Dv2_23252952941219
// MI455X (gfx1250) — matrix operand order repaired and checked by reading, not compiled, not run on hardware
//
#include <hip/hip_runtime.h>
#include <hip/hip_bf16.h>
#include <stdint.h>

#define NREL 67
#define NPAD 80
#define NVOX 2025
#define NPADJ 2048  /* padded row stride of xb so 32-j chunks are 16B aligned */
#define FEAT 256
#define HALF 128
#define KCAT 8960   /* 256 + 67*128 + 128 */
#define CTX0 256    /* Y row offset of x_context */
#define XROW 8832   /* Y row offset of x */

#if defined(__has_builtin)
#  if __has_builtin(__builtin_amdgcn_global_load_async_to_lds_b128) && \
      __has_builtin(__builtin_amdgcn_s_wait_asynccnt)
#    define USE_ASYNC_LDS 1
#  endif
#endif
#ifndef USE_ASYNC_LDS
#  define USE_ASYNC_LDS 0
#endif

typedef __attribute__((ext_vector_type(16))) __bf16 v16bf;
typedef __attribute__((ext_vector_type(8)))  float  v8f;

#if USE_ASYNC_LDS
typedef int v4i __attribute__((__vector_size__(4 * sizeof(int))));
typedef __attribute__((address_space(1))) v4i* gv4i_p;   // global int4*
typedef __attribute__((address_space(3))) v4i* lv4i_p;   // LDS int4*
#endif

__device__ __forceinline__ __bf16 f2bf(float f) {
  unsigned u = __builtin_bit_cast(unsigned, f);
  unsigned r = (u + 0x7FFFu + ((u >> 16) & 1u)) >> 16;  // RNE to bf16
  return __builtin_bit_cast(__bf16, (unsigned short)r);
}

// ---------- small VALU kernels (negligible FLOPs) ----------

__global__ __launch_bounds__(256) void k_agg(const float* __restrict__ in,
    const float* __restrict__ w, const float* __restrict__ b, float* __restrict__ out) {
  int idx = blockIdx.x * 256 + threadIdx.x;
  if (idx >= HALF * NVOX) return;
  int o = idx / NVOX, n = idx % NVOX;
  float s = b[o];
  const float* wr = w + (size_t)o * FEAT;
  for (int c = 0; c < FEAT; ++c) s += wr[c] * in[(size_t)c * NVOX + n];
  out[idx] = s;
}

__global__ __launch_bounds__(256) void k_reduce(const float* __restrict__ x,
    const float* __restrict__ w, const float* __restrict__ g, const float* __restrict__ b,
    float* __restrict__ out) {
  int idx = blockIdx.x * 256 + threadIdx.x;
  if (idx >= 32 * NVOX) return;
  int o = idx / NVOX, n = idx % NVOX;
  float s = 0.f;
  const float* wr = w + (size_t)o * HALF;
  for (int c = 0; c < HALF; ++c) s += wr[c] * x[(size_t)c * NVOX + n];
  out[idx] = fmaxf(g[o] * s + b[o], 0.f);
}

// 3-tap dilated conv along one axis of the (15,15,9) grid + BN (+opt input relu, +opt residual)
__global__ __launch_bounds__(256) void k_axis(const float* __restrict__ in,
    const float* __restrict__ w, const float* __restrict__ g, const float* __restrict__ b,
    const float* __restrict__ resid, float* __restrict__ out,
    int stride, int extent, int dil, int relu_in) {
  int idx = blockIdx.x * 256 + threadIdx.x;
  if (idx >= 32 * NVOX) return;
  int o = idx / NVOX, n = idx % NVOX;
  int coord = (n / stride) % extent;
  int off = dil * stride;
  float s = 0.f;
  for (int i = 0; i < 32; ++i) {
    const float* wi = w + ((size_t)o * 32 + i) * 3;
    float v0 = (coord - dil >= 0)     ? in[(size_t)i * NVOX + n - off] : 0.f;
    float v1 = in[(size_t)i * NVOX + n];
    float v2 = (coord + dil < extent) ? in[(size_t)i * NVOX + n + off] : 0.f;
    if (relu_in) { v0 = fmaxf(v0, 0.f); v1 = fmaxf(v1, 0.f); v2 = fmaxf(v2, 0.f); }
    s += wi[0] * v0 + wi[1] * v1 + wi[2] * v2;
  }
  float r = g[o] * s + b[o];
  if (resid) r += resid[idx];
  out[idx] = r;
}

__global__ __launch_bounds__(256) void k_expand(const float* __restrict__ t,
    const float* __restrict__ w, const float* __restrict__ g, const float* __restrict__ b,
    const float* __restrict__ xold, float* __restrict__ xnew) {
  int idx = blockIdx.x * 256 + threadIdx.x;
  if (idx >= HALF * NVOX) return;
  int o = idx / NVOX, n = idx % NVOX;
  float s = 0.f;
  const float* wr = w + (size_t)o * 32;
  for (int c = 0; c < 32; ++c) s += wr[c] * fmaxf(t[(size_t)c * NVOX + n], 0.f);
  xnew[idx] = fmaxf(g[o] * s + b[o] + xold[idx], 0.f);
}

__global__ __launch_bounds__(256) void k_amac(const float* __restrict__ x,
    const float* __restrict__ pw, float* __restrict__ am, float* __restrict__ ac) {
  int idx = blockIdx.x * 256 + threadIdx.x;
  if (idx >= NREL * NVOX) return;
  int r = idx / NVOX, n = idx % NVOX;
  const float* wm = pw + (size_t)r * FEAT;
  const float* wc = wm + HALF;
  float sm = 0.f, sc = 0.f;
  for (int c = 0; c < HALF; ++c) {
    float xv = x[(size_t)c * NVOX + n];
    sm += wm[c] * xv; sc += wc[c] * xv;
  }
  am[idx] = sm; ac[idx] = sc;
}

__global__ __launch_bounds__(256) void k_pack_in(const float* __restrict__ in, __bf16* __restrict__ Y) {
  int idx = blockIdx.x * 256 + threadIdx.x;
  if (idx < FEAT * NVOX) Y[idx] = f2bf(in[idx]);
}

// pack x into (a) Y rows [XROW..XROW+128) and (b) c-major padded bf16 matrix xb[128][NPADJ]
__global__ __launch_bounds__(256) void k_pack_x(const float* __restrict__ x,
    __bf16* __restrict__ Y, __bf16* __restrict__ xb) {
  int idx = blockIdx.x * 256 + threadIdx.x;
  if (idx >= HALF * NPADJ) return;
  int c = idx / NPADJ, n = idx % NPADJ;
  __bf16 v = f2bf(0.f);
  if (n < NVOX) {
    v = f2bf(x[(size_t)c * NVOX + n]);
    Y[(size_t)(XROW + c) * NVOX + n] = v;
  }
  xb[idx] = v;  // zero padding beyond NVOX
}

__global__ __launch_bounds__(256) void k_pack_w(const float* __restrict__ w, __bf16* __restrict__ Wb) {
  int idx = blockIdx.x * 256 + threadIdx.x;
  if (idx < FEAT * KCAT) Wb[idx] = f2bf(w[idx]);
}

// ---------- fused prior kernel: logits -> P_logit store -> softmax -> masked ctx GEMM ----------
// One workgroup (8 waves) per voxel i. Accumulates ctx[80(pad67) x 128] in registers
// across all 2025 j's with v_wmma_f32_16x16x32_bf16. Single pass over the 1.1 GB P_logit.
__global__ __launch_bounds__(256) void k_prior(
    const float* __restrict__ am, const float* __restrict__ ac,
    const float* __restrict__ predb, const int* __restrict__ masks,
    const __bf16* __restrict__ xb, float* __restrict__ plog, __bf16* __restrict__ Y) {
  const int i = blockIdx.x;
  const int tid = threadIdx.x;
  const int lane = tid & 31, wave = tid >> 5;
  const int half = lane >> 4, nl = lane & 15;
  const int nb = wave * 16;  // this wave's 16-column slice of the 128 channels

  __shared__ float  sAi[NREL];
  __shared__ float  sP[NREL][32];
  __shared__ __bf16 sA[NPAD][32];     // A tile: masked probs, K contiguous per row
  __shared__ __bf16 sBk[HALF][40];    // B tile transposed: [c][k], padded stride (80B rows)

  if (tid < NREL) sAi[tid] = am[(size_t)tid * NVOX + i] + predb[tid];
  for (int v = tid; v < (NPAD - NREL) * 32; v += 256)
    sA[NREL + v / 32][v % 32] = f2bf(0.f);
  const int mi = masks[i];

  v8f acc[5];
#pragma unroll
  for (int rb = 0; rb < 5; ++rb)
#pragma unroll
    for (int e = 0; e < 8; ++e) acc[rb][e] = 0.f;

  __syncthreads();

  for (int jc = 0; jc < (NVOX + 31) / 32; ++jc) {
    const int j0 = jc * 32;
    // prefetch next chunk's logits operand (global_prefetch_b8)
    if (j0 + 32 < NVOX) {
      if (tid < NREL) __builtin_prefetch(&ac[(size_t)tid * NVOX + j0 + 32], 0, 3);
#if !USE_ASYNC_LDS
      if (tid < HALF) __builtin_prefetch(&xb[(size_t)tid * NPADJ + j0 + 32], 0, 3);
#endif
    }
    // A: raw logits into LDS
    for (int e = tid; e < NREL * 32; e += 256) {
      int r = e / 32, jj = e & 31, j = j0 + jj;
      sP[r][jj] = (j < NVOX) ? (sAi[r] + ac[(size_t)r * NVOX + j]) : 0.f;
    }
    __syncthreads();
    // B: stage B tile (async memory->LDS when available), stream P_logit chunk out
#if USE_ASYNC_LDS
    for (int e = tid; e < HALF * 4; e += 256) {   // 512 x b128 = 8 KB tile, 16B aligned both sides
      int c = e >> 2, q = e & 3;
      __builtin_amdgcn_global_load_async_to_lds_b128(
          (gv4i_p)(uintptr_t)(xb + (size_t)c * NPADJ + j0 + q * 8),
          (lv4i_p)(uint32_t)(uintptr_t)(&sBk[c][q * 8]),
          0, 0);
    }
#else
    for (int e = tid; e < HALF * 16; e += 256) {
      int c = e >> 4, kw = e & 15;
      const unsigned* src = (const unsigned*)(xb + (size_t)c * NPADJ + j0);
      unsigned* dst = (unsigned*)(&sBk[c][0]);
      dst[kw] = src[kw];
    }
#endif
    for (int e = tid; e < NREL * 32; e += 256) {
      int r = e / 32, jj = e & 31, j = j0 + jj;
      if (j < NVOX)
        plog[(size_t)r * NVOX * NVOX + (size_t)i * NVOX + j] = sP[r][jj];
    }
#if USE_ASYNC_LDS
    __builtin_amdgcn_s_wait_asynccnt(0);
#endif
    __syncthreads();
    // C: softmax over 67 relations per column (T=0.03); 8 threads per column
    {
      int jj = tid >> 3, sub = tid & 7;
      int j = j0 + jj;
      if (j < NVOX) {
        float mx = -3.0e38f;
        for (int r = sub; r < NREL; r += 8) mx = fmaxf(mx, sP[r][jj]);
        mx = fmaxf(mx, __shfl_xor(mx, 1));
        mx = fmaxf(mx, __shfl_xor(mx, 2));
        mx = fmaxf(mx, __shfl_xor(mx, 4));
        const float is = 1.f / 0.03f;
        float sum = 0.f;
        for (int r = sub; r < NREL; r += 8) {
          float e = __expf((sP[r][jj] - mx) * is);
          sP[r][jj] = e; sum += e;
        }
        sum += __shfl_xor(sum, 1);
        sum += __shfl_xor(sum, 2);
        sum += __shfl_xor(sum, 4);
        float m = (mi || masks[j]) ? 1.f : 0.f;
        float sc = m / sum;
        for (int r = sub; r < NREL; r += 8) sA[r][jj] = f2bf(sP[r][jj] * sc);
      } else {
        for (int r = sub; r < NREL; r += 8) sA[r][jj] = f2bf(0.f);
      }
    }
    __syncthreads();
    // D: 5 WMMAs per wave (r-blocks 0..4 x this wave's n-block), K = 32 j's
    v16bf bfr;
#pragma unroll
    for (int t = 0; t < 8; ++t) {             // B layout: VGPR t -> K = half*16 + 2t (contig -> b128)
      int k = half * 16 + 2 * t;
      bfr[2 * t]     = sBk[nb + nl][k];
      bfr[2 * t + 1] = sBk[nb + nl][k + 1];
    }
#pragma unroll
    for (int rb = 0; rb < 5; ++rb) {
      v16bf afr;
      int M = rb * 16 + nl;
#pragma unroll
      for (int t = 0; t < 8; ++t) {           // A layout: VGPR t -> K = half*16 + 2t (matches B)
        int k = half * 16 + 2 * t;
        afr[2 * t]     = sA[M][k];
        afr[2 * t + 1] = sA[M][k + 1];
      }
      acc[rb] = __builtin_amdgcn_wmma_f32_16x16x32_bf16(
          false, afr, false, bfr, (short)0, acc[rb], false, false);
    }
    __syncthreads();
  }

  // write ctx (bf16) straight into Y rows [256, 256+8576): Y[256 + r*128 + c][i]
#pragma unroll
  for (int rb = 0; rb < 5; ++rb)
#pragma unroll
    for (int v = 0; v < 8; ++v) {
      int r = rb * 16 + v + 8 * half;         // C/D layout: VGPR v, lanes16-31 -> M+8
      if (r < NREL) {
        int c = nb + nl;
        Y[(size_t)(CTX0 + r * HALF + c) * NVOX + i] = f2bf(acc[rb][v]);
      }
    }
}

// ---------- resize: out[256 x 2025] = Wb[256 x 8960] @ Y[8960 x 2025] + bias ----------
__global__ __launch_bounds__(256) void k_resize(
    const __bf16* __restrict__ Wb, const __bf16* __restrict__ Y,
    const float* __restrict__ bias, float* __restrict__ out) {
  const int tid = threadIdx.x;
  const int lane = tid & 31, wave = tid >> 5;
  const int half = lane >> 4, nl = lane & 15;
  const int i0 = blockIdx.x * 16;

  __shared__ __bf16 sNK[16][40];  // B tile transposed: [n][k], padded rows
  v8f acc[2];
#pragma unroll
  for (int mt = 0; mt < 2; ++mt)
#pragma unroll
    for (int e = 0; e < 8; ++e) acc[mt][e] = 0.f;

  for (int kc = 0; kc < KCAT / 32; ++kc) {
    const int k0 = kc * 32;
    if (tid < 32 && kc + 1 < KCAT / 32)
      __builtin_prefetch(&Y[(size_t)(k0 + 32 + tid) * NVOX + i0], 0, 3);
    for (int e = tid; e < 32 * 16; e += 256) {
      int k = e >> 4, n = e & 15, j = i0 + n;
      sNK[n][k] = (j < NVOX) ? Y[(size_t)(k0 + k) * NVOX + j] : f2bf(0.f);
    }
    __syncthreads();
    v16bf bfr;
#pragma unroll
    for (int t = 0; t < 8; ++t) {             // contiguous per lane -> ds_load_b128
      int k = half * 16 + 2 * t;
      bfr[2 * t]     = sNK[nl][k];
      bfr[2 * t + 1] = sNK[nl][k + 1];
    }
#pragma unroll
    for (int mt = 0; mt < 2; ++mt) {
      int m0 = (wave * 2 + mt) * 16;
      const __bf16* wrow = Wb + (size_t)(m0 + nl) * KCAT + k0;
      v16bf afr;
#pragma unroll
      for (int t = 0; t < 8; ++t) {           // one contiguous 32B run per lane -> global_load_b128 x2
        int k = half * 16 + 2 * t;
        afr[2 * t]     = wrow[k];
        afr[2 * t + 1] = wrow[k + 1];
      }
      acc[mt] = __builtin_amdgcn_wmma_f32_16x16x32_bf16(
          false, afr, false, bfr, (short)0, acc[mt], false, false);
    }
    __syncthreads();
  }
#pragma unroll
  for (int mt = 0; mt < 2; ++mt)
#pragma unroll
    for (int v = 0; v < 8; ++v) {
      int o = (wave * 2 + mt) * 16 + v + 8 * half;
      int ii = i0 + nl;
      if (ii < NVOX) out[(size_t)o * NVOX + ii] = acc[mt][v] + bias[o];
    }
}

extern "C" void kernel_launch(void* const* d_in, const int* in_sizes, int n_in,
                              void* d_out, int out_size, void* d_ws, size_t ws_size,
                              hipStream_t stream) {
  (void)in_sizes; (void)n_in; (void)out_size; (void)ws_size;
  const float* input  = (const float*)d_in[0];
  const int*   masks  = (const int*)d_in[1];
  const float* agg_w  = (const float*)d_in[2];
  const float* agg_b  = (const float*)d_in[3];
  const float* pred_w = (const float*)d_in[4];
  const float* pred_b = (const float*)d_in[5];
  const float* rs_w   = (const float*)d_in[6];
  const float* rs_b   = (const float*)d_in[7];

  float* out  = (float*)d_out;
  float* plog = out + (size_t)FEAT * NVOX;   // outputs concatenated: (out, P_logit)

  char* wp = (char*)d_ws;
  auto alloc = [&](size_t bytes) -> void* {
    void* p = (void*)wp;
    wp += (bytes + 255) & ~(size_t)255;
    return p;
  };
  float*  x0 = (float*)alloc((size_t)HALF * NVOX * 4);
  float*  x1 = (float*)alloc((size_t)HALF * NVOX * 4);
  float*  t1 = (float*)alloc((size_t)32 * NVOX * 4);
  float*  t2 = (float*)alloc((size_t)32 * NVOX * 4);
  float*  t3 = (float*)alloc((size_t)32 * NVOX * 4);
  float*  t4 = (float*)alloc((size_t)32 * NVOX * 4);
  float*  am = (float*)alloc((size_t)NREL * NVOX * 4);
  float*  ac = (float*)alloc((size_t)NREL * NVOX * 4);
  __bf16* xb = (__bf16*)alloc((size_t)HALF * NPADJ * 2);
  __bf16* Wb = (__bf16*)alloc((size_t)FEAT * KCAT * 2);
  __bf16* Y  = (__bf16*)alloc((size_t)KCAT * NVOX * 2);

  auto nb = [](int n) { return (n + 255) / 256; };

  k_agg<<<nb(HALF * NVOX), 256, 0, stream>>>(input, agg_w, agg_b, x0);

  float* xin = x0; float* xout = x1;
  for (int bt = 0; bt < 3; ++bt) {
    const int dil = bt + 1;
    const int p0 = 8 + bt * 15;
    const float* w1 = (const float*)d_in[p0 + 0];
    const float* g1 = (const float*)d_in[p0 + 1];
    const float* b1 = (const float*)d_in[p0 + 2];
    const float* w2 = (const float*)d_in[p0 + 3];
    const float* g2 = (const float*)d_in[p0 + 4];
    const float* b2 = (const float*)d_in[p0 + 5];
    const float* w3 = (const float*)d_in[p0 + 6];
    const float* g3 = (const float*)d_in[p0 + 7];
    const float* b3 = (const float*)d_in[p0 + 8];
    const float* w4 = (const float*)d_in[p0 + 9];
    const float* g4 = (const float*)d_in[p0 + 10];
    const float* b4 = (const float*)d_in[p0 + 11];
    const float* w5 = (const float*)d_in[p0 + 12];
    const float* g5 = (const float*)d_in[p0 + 13];
    const float* b5 = (const float*)d_in[p0 + 14];

    k_reduce<<<nb(32 * NVOX), 256, 0, stream>>>(xin, w1, g1, b1, t1);
    k_axis<<<nb(32 * NVOX), 256, 0, stream>>>(t1, w2, g2, b2, nullptr, t2, 1, 9, dil, 0);
    k_axis<<<nb(32 * NVOX), 256, 0, stream>>>(t2, w3, g3, b3, t2, t3, 9, 15, dil, 1);
    k_axis<<<nb(32 * NVOX), 256, 0, stream>>>(t3, w4, g4, b4, t3, t4, 135, 15, dil, 1);
    k_expand<<<nb(HALF * NVOX), 256, 0, stream>>>(t4, w5, g5, b5, xin, xout);
    float* tmp = xin; xin = xout; xout = tmp;
  }

  k_amac<<<nb(NREL * NVOX), 256, 0, stream>>>(xin, pred_w, am, ac);
  k_pack_in<<<nb(FEAT * NVOX), 256, 0, stream>>>(input, Y);
  k_pack_x<<<nb(HALF * NPADJ), 256, 0, stream>>>(xin, Y, xb);
  k_pack_w<<<nb(FEAT * KCAT), 256, 0, stream>>>(rs_w, Wb);

  k_prior<<<NVOX, 256, 0, stream>>>(am, ac, pred_b, masks, xb, plog, Y);
  k_resize<<<(NVOX + 15) / 16, 256, 0, stream>>>(Wb, Y, rs_b, out);
}
